// SwinTransformerFusion_83915071029400
// MI455X (gfx1250) — matrix operand order repaired and checked by reading, not compiled, not run on hardware
//
#include <hip/hip_runtime.h>
#include <hip/hip_bf16.h>

// MI455X / gfx1250, wave32, WMMA bf16 pipeline for the Swin fusion block.
// Workspace requirement: ~274 MB (see offsets below).

typedef __bf16 bf16_t;
typedef __attribute__((ext_vector_type(16))) __bf16 v16bf;
typedef __attribute__((ext_vector_type(8)))  float  v8f;

#define WMMA_BF16(a, b, c) \
  __builtin_amdgcn_wmma_f32_16x16x32_bf16(false, (a), false, (b), (short)0, (c), false, false)

// ---------------- constants ----------------
// B=8, DIM=192, C2=384, H=W=60, HP=WP=63, WS=7, SHIFT=3
// NW=81 windows/batch, N=49 tokens/window, HEADS=4, HD=96, MLP_H=1536
#define NB       8
#define C2       384
#define HEADS    4
#define HD       96
#define MLPH     1536
#define HPIX     60
#define HPAD     63
#define NTOK     31752          // 8*63*63 == 8*81*49
#define NWIN     648            // 8*81
#define CONVM    28800          // 8*60*60
#define CONVK    3456           // 9*384

union Frag { v16bf v; bf16_t e[16]; uint4 q[2]; };

// A-matrix fragment (16x32, 16-bit): lane<16: K=k0+0..15;
// lane>=16: K=k0+16..31. Two 16-byte loads.
__device__ inline Frag load_frag_A(const bf16_t* base, int ld, int row, int k0, int lane) {
  Frag f;
  const bf16_t* p = base + (size_t)row * ld + k0 + ((lane >> 4) << 4);
  f.q[0] = *(const uint4*)(p);
  f.q[1] = *(const uint4*)(p + 8);
  return f;
}

// B-matrix fragment (32x16, 16-bit): lane holds column n, elements are
// K = k0 + 16*(lane>=16) + e  -> 32 contiguous bytes of W(n, k).
__device__ inline Frag load_frag_B(const bf16_t* base, int ld, int row, int k0, int lane) {
  Frag f;
  const bf16_t* p = base + (size_t)row * ld + k0 + ((lane >> 4) << 4);
  f.q[0] = *(const uint4*)(p);
  f.q[1] = *(const uint4*)(p + 8);
  return f;
}

__device__ inline float wave_sum(float v) {
#pragma unroll
  for (int o = 16; o >= 1; o >>= 1) v += __shfl_xor(v, o, 32);
  return v;
}

// ---------------- weight prep ----------------
__global__ __launch_bounds__(256) void cvt_f32_bf16(const float* __restrict__ s,
                                                    bf16_t* __restrict__ d, int n) {
  int i = blockIdx.x * 256 + threadIdx.x;
  if (i < n) d[i] = (bf16_t)s[i];
}

// conv_w (O=192, I=384, 3, 3) -> (O, tap*384 + c), tap = kh*3+kw
__global__ __launch_bounds__(256) void conv_repack(const float* __restrict__ s,
                                                   bf16_t* __restrict__ d) {
  int i = blockIdx.x * 256 + threadIdx.x;
  if (i >= 192 * CONVK) return;
  int o = i / CONVK, r = i - o * CONVK;
  int tap = r / 384, c = r - tap * 384;
  d[i] = (bf16_t)s[(size_t)o * CONVK + c * 9 + tap];
}

// ---------------- concat + pad + LN(norm) -> z (f32, spatial NTOKxC2) ----------------
__global__ __launch_bounds__(256) void concat_ln_kernel(const float* __restrict__ x,
                                                        const float* __restrict__ g,
                                                        const float* __restrict__ w,
                                                        const float* __restrict__ bb,
                                                        float* __restrict__ z) {
  int wid = threadIdx.x >> 5, lane = threadIdx.x & 31;
  int tok = blockIdx.x * 8 + wid;
  int b = tok / (HPAD * HPAD), rem = tok - b * (HPAD * HPAD);
  int h = rem / HPAD, ww = rem - h * HPAD;
  bool inb = (h < HPIX) && (ww < HPIX);
  float v[12];
  float s = 0.f;
#pragma unroll
  for (int e = 0; e < 12; ++e) {
    int c = lane + 32 * e;
    float t = 0.f;
    if (inb) {
      if (c < 192) t = x[((size_t)(b * 192 + c) * HPIX + h) * HPIX + ww];
      else         t = g[((size_t)(b * 192 + (c - 192)) * HPIX + h) * HPIX + ww];
    }
    v[e] = t; s += t;
  }
  float mu = wave_sum(s) * (1.f / 384.f);
  float q = 0.f;
#pragma unroll
  for (int e = 0; e < 12; ++e) { float d = v[e] - mu; q += d * d; }
  float var = wave_sum(q) * (1.f / 384.f);
  float rs = rsqrtf(var + 1e-5f);
#pragma unroll
  for (int e = 0; e < 12; ++e) {
    int c = lane + 32 * e;
    z[(size_t)tok * C2 + c] = (v[e] - mu) * rs * w[c] + bb[c];
  }
}

// LN of z row -> bf16 out.  permute=1: output index is the *windowed* token
// (roll -3 then 7x7 partition), source is spatial.  permute=0: identity order.
__global__ __launch_bounds__(256) void ln_bf16_kernel(const float* __restrict__ z,
                                                      const float* __restrict__ w,
                                                      const float* __restrict__ bb,
                                                      bf16_t* __restrict__ out,
                                                      int permute) {
  int wid = threadIdx.x >> 5, lane = threadIdx.x & 31;
  int tok = blockIdx.x * 8 + wid;
  int src = tok;
  if (permute) {
    int winI = tok / 49, n = tok - winI * 49;
    int b = winI / 81, rr = winI - b * 81;
    int hb = rr / 9, wb = rr - hb * 9;
    int i = n / 7, j = n - i * 7;
    int h = hb * 7 + i + 3; if (h >= HPAD) h -= HPAD;
    int ww = wb * 7 + j + 3; if (ww >= HPAD) ww -= HPAD;
    src = (b * HPAD + h) * HPAD + ww;
  }
  float v[12];
  float s = 0.f;
#pragma unroll
  for (int e = 0; e < 12; ++e) {
    v[e] = z[(size_t)src * C2 + lane + 32 * e]; s += v[e];
  }
  float mu = wave_sum(s) * (1.f / 384.f);
  float q = 0.f;
#pragma unroll
  for (int e = 0; e < 12; ++e) { float d = v[e] - mu; q += d * d; }
  float var = wave_sum(q) * (1.f / 384.f);
  float rs = rsqrtf(var + 1e-5f);
#pragma unroll
  for (int e = 0; e < 12; ++e) {
    int c = lane + 32 * e;
    out[(size_t)tok * C2 + c] = (bf16_t)((v[e] - mu) * rs * w[c] + bb[c]);
  }
}

// ---------------- generic bf16 WMMA GEMM:  C = A(MxK) * W(NxK)^T ----------------
// mode 0: out = bf16(c + bias)                       (QKV)
// mode 1: out = bf16(gelu(c + bias))                 (fc1)
// mode 2: z[spatial(row)] += c + bias  (in-place)    (proj + window-reverse + unshift + residual)
// mode 3: out = bf16(z2[row] + c + bias)             (fc2 + residual)
__global__ __launch_bounds__(256) void gemm_bf16_kernel(const bf16_t* __restrict__ A,
                                                        const bf16_t* __restrict__ W,
                                                        const float* __restrict__ bias,
                                                        bf16_t* __restrict__ outBF,
                                                        float* __restrict__ zbuf,
                                                        int M, int N, int K, int mode) {
  int tid = threadIdx.x;
  int lane = tid & 31, wid = tid >> 5;
  int l15 = lane & 15, half = lane >> 4;
  int waveM = wid & 3, waveN = wid >> 2;
  int mBase = blockIdx.x * 128 + waveM * 32;
  int nBase = blockIdx.y * 64 + waveN * 32;

  int ra0 = mBase + l15;      if (ra0 > M - 1) ra0 = M - 1;
  int ra1 = mBase + 16 + l15; if (ra1 > M - 1) ra1 = M - 1;
  int rb0 = nBase + l15;
  int rb1 = nBase + 16 + l15;

  v8f acc[2][2] = {};
  for (int k0 = 0; k0 < K; k0 += 32) {
    Frag a0 = load_frag_A(A, K, ra0, k0, lane);
    Frag a1 = load_frag_A(A, K, ra1, k0, lane);
    Frag b0 = load_frag_B(W, K, rb0, k0, lane);
    Frag b1 = load_frag_B(W, K, rb1, k0, lane);
    acc[0][0] = WMMA_BF16(a0.v, b0.v, acc[0][0]);
    acc[0][1] = WMMA_BF16(a0.v, b1.v, acc[0][1]);
    acc[1][0] = WMMA_BF16(a1.v, b0.v, acc[1][0]);
    acc[1][1] = WMMA_BF16(a1.v, b1.v, acc[1][1]);
  }

#pragma unroll
  for (int mi = 0; mi < 2; ++mi)
#pragma unroll
    for (int ni = 0; ni < 2; ++ni)
#pragma unroll
      for (int r = 0; r < 8; ++r) {
        int row = mBase + mi * 16 + r + 8 * half;
        int col = nBase + ni * 16 + l15;
        if (row >= M) continue;
        float c = acc[mi][ni][r] + bias[col];
        if (mode == 0) {
          outBF[(size_t)row * N + col] = (bf16_t)c;
        } else if (mode == 1) {
          outBF[(size_t)row * N + col] =
              (bf16_t)(0.5f * c * (1.0f + erff(c * 0.70710678118654752f)));
        } else if (mode == 2) {
          int winI = row / 49, n = row - winI * 49;
          int b = winI / 81, rr = winI - b * 81;
          int hb = rr / 9, wb = rr - hb * 9;
          int i = n / 7, j = n - i * 7;
          int h = hb * 7 + i + 3; if (h >= HPAD) h -= HPAD;
          int w2 = wb * 7 + j + 3; if (w2 >= HPAD) w2 -= HPAD;
          size_t sp = ((size_t)(b * HPAD + h) * HPAD + w2) * C2 + col;
          zbuf[sp] += c;
        } else {  // mode 3
          float v = zbuf[(size_t)row * C2 + col] + c;
          outBF[(size_t)row * C2 + col] = (bf16_t)v;
        }
      }
}

// ---------------- attention: per (window, head) workgroup, 4 waves ----------------
// qkv layout: row = win*49 + token, cols: q[h*96+d] | 384+k | 768+v
__global__ __launch_bounds__(128) void attn_kernel(const bf16_t* __restrict__ qkv,
                                                   const float* __restrict__ rpb,
                                                   bf16_t* __restrict__ out) {
  int blk = blockIdx.x;
  int win = blk >> 2;
  int head = blk & 3;
  int tid = threadIdx.x;
  int lane = tid & 31, wid = tid >> 5;  // wid = row-tile (0..3)
  int l15 = lane & 15, half = lane >> 4;

  __shared__ float  sS[64 * 64];
  __shared__ bf16_t sP[64 * 64];

  size_t wbase = (size_t)win * 49 * 1152;
  const bf16_t* qb = qkv + wbase + head * HD;
  const bf16_t* kb = qkv + wbase + 384 + head * HD;
  const bf16_t* vb = qkv + wbase + 768 + head * HD;

  // ---- S = Q * K^T ----
  v8f acc[4] = {};
  for (int ks = 0; ks < HD; ks += 32) {
    int arow = wid * 16 + l15; if (arow > 48) arow = 48;
    Frag fa = load_frag_A(qb, 1152, arow, ks, lane);
#pragma unroll
    for (int ct = 0; ct < 4; ++ct) {
      int brow = ct * 16 + l15; if (brow > 48) brow = 48;
      Frag fb = load_frag_B(kb, 1152, brow, ks, lane);
      acc[ct] = WMMA_BF16(fa.v, fb.v, acc[ct]);
    }
  }

  // scale + rel-pos bias + shift mask, store to LDS
  int r81 = win % 81;
  int hb = r81 / 9, wb = r81 - hb * 9;
#pragma unroll
  for (int ct = 0; ct < 4; ++ct)
#pragma unroll
    for (int r = 0; r < 8; ++r) {
      int n = wid * 16 + r + 8 * half;
      int m = ct * 16 + l15;
      float v = acc[ct][r] * 0.1020620726f;  // 96^-0.5
      if (n < 49 && m < 49) {
        int i1 = n / 7, j1 = n - i1 * 7, i2 = m / 7, j2 = m - i2 * 7;
        int rel = (i1 - i2 + 6) * 13 + (j1 - j2 + 6);
        v += rpb[rel * HEADS + head];
        int hA = hb * 7 + i1, wA = wb * 7 + j1;
        int hB = hb * 7 + i2, wB = wb * 7 + j2;
        int ra = (hA < 56 ? 0 : (hA < 60 ? 1 : 2)) * 3 + (wA < 56 ? 0 : (wA < 60 ? 1 : 2));
        int rb = (hB < 56 ? 0 : (hB < 60 ? 1 : 2)) * 3 + (wB < 56 ? 0 : (wB < 60 ? 1 : 2));
        if (ra != rb) v -= 100.0f;
      }
      sS[n * 64 + m] = v;
    }
  __syncthreads();

  // ---- softmax rows (49 valid), write P bf16 with zero padding ----
  if (tid < 49) {
    float mx = -1e30f;
    for (int m = 0; m < 49; ++m) mx = fmaxf(mx, sS[tid * 64 + m]);
    float s = 0.f;
    for (int m = 0; m < 49; ++m) s += __expf(sS[tid * 64 + m] - mx);
    float inv = 1.0f / s;
    for (int m = 0; m < 64; ++m)
      sP[tid * 64 + m] = (m < 49) ? (bf16_t)(__expf(sS[tid * 64 + m] - mx) * inv)
                                  : (bf16_t)0.0f;
  } else if (tid < 64) {
    for (int m = 0; m < 64; ++m) sP[tid * 64 + m] = (bf16_t)0.0f;
  }
  __syncthreads();

  // ---- O = P * V ----
  v8f o[6] = {};
  for (int ks = 0; ks < 64; ks += 32) {
    Frag fa = load_frag_A((const bf16_t*)sP, 64, wid * 16 + l15, ks, lane);
#pragma unroll
    for (int ct = 0; ct < 6; ++ct) {
      Frag fb;
      int d = ct * 16 + l15;
#pragma unroll
      for (int e = 0; e < 16; ++e) {
        int m = ks + e + 16 * half;   // B fragment: K = k0 + 16*half + e
        if (m > 48) m = 48;           // P is zero there anyway
        fb.e[e] = vb[(size_t)m * 1152 + d];
      }
      o[ct] = WMMA_BF16(fa.v, fb.v, o[ct]);
    }
  }
#pragma unroll
  for (int ct = 0; ct < 6; ++ct)
#pragma unroll
    for (int r = 0; r < 8; ++r) {
      int n = wid * 16 + r + 8 * half;
      if (n < 49)
        out[((size_t)win * 49 + n) * C2 + head * HD + ct * 16 + l15] = (bf16_t)o[ct][r];
    }
}

// ---------------- implicit-GEMM 3x3 conv + residual x -> d_out ----------------
__device__ inline Frag load_frag_conv(const bf16_t* __restrict__ z3, int p, int k0, int lane) {
  int b = p / 3600, rem = p - b * 3600;
  int h = rem / 60, w = rem - h * 60;
  int half = lane >> 4;
  Frag f;
#pragma unroll
  for (int gq = 0; gq < 2; ++gq) {
    int kk = k0 + half * 16 + gq * 8;  // A-fragment k offsets
    int tap = kk / 384, c = kk - tap * 384;
    int hh = h + tap / 3 - 1;
    int ww = w + tap % 3 - 1;
    uint4 val = {0u, 0u, 0u, 0u};
    if (hh >= 0 && hh < HPIX && ww >= 0 && ww < HPIX)
      val = *(const uint4*)(z3 + ((size_t)((b * HPAD + hh) * HPAD + ww)) * C2 + c);
    f.q[gq] = val;
  }
  return f;
}

__global__ __launch_bounds__(256) void conv_kernel(const bf16_t* __restrict__ z3,
                                                   const bf16_t* __restrict__ Wc,
                                                   const float* __restrict__ cb,
                                                   const float* __restrict__ x,
                                                   float* __restrict__ out) {
  int tid = threadIdx.x;
  int lane = tid & 31, wid = tid >> 5;
  int l15 = lane & 15, half = lane >> 4;
  int waveM = wid & 3, waveN = wid >> 2;
  int mBase = blockIdx.x * 128 + waveM * 32;
  int nBase = blockIdx.y * 64 + waveN * 32;

  int p0 = mBase + l15;      if (p0 > CONVM - 1) p0 = CONVM - 1;
  int p1 = mBase + 16 + l15; if (p1 > CONVM - 1) p1 = CONVM - 1;

  v8f acc[2][2] = {};
  for (int k0 = 0; k0 < CONVK; k0 += 32) {
    Frag a0 = load_frag_conv(z3, p0, k0, lane);
    Frag a1 = load_frag_conv(z3, p1, k0, lane);
    Frag b0 = load_frag_B(Wc, CONVK, nBase + l15, k0, lane);
    Frag b1 = load_frag_B(Wc, CONVK, nBase + 16 + l15, k0, lane);
    acc[0][0] = WMMA_BF16(a0.v, b0.v, acc[0][0]);
    acc[0][1] = WMMA_BF16(a0.v, b1.v, acc[0][1]);
    acc[1][0] = WMMA_BF16(a1.v, b0.v, acc[1][0]);
    acc[1][1] = WMMA_BF16(a1.v, b1.v, acc[1][1]);
  }

#pragma unroll
  for (int mi = 0; mi < 2; ++mi)
#pragma unroll
    for (int ni = 0; ni < 2; ++ni)
#pragma unroll
      for (int r = 0; r < 8; ++r) {
        int p = mBase + mi * 16 + r + 8 * half;
        if (p >= CONVM) continue;
        int o = nBase + ni * 16 + l15;
        int b = p / 3600, rem = p - b * 3600;
        int h = rem / 60, w = rem - h * 60;
        size_t idx = ((size_t)(b * 192 + o) * HPIX + h) * HPIX + w;
        out[idx] = acc[mi][ni][r] + cb[o] + x[idx];
      }
}

// ---------------- launch ----------------
extern "C" void kernel_launch(void* const* d_in, const int* in_sizes, int n_in,
                              void* d_out, int out_size, void* d_ws, size_t ws_size,
                              hipStream_t stream) {
  (void)in_sizes; (void)n_in; (void)out_size; (void)ws_size;
  const float* x      = (const float*)d_in[0];
  const float* g      = (const float*)d_in[1];
  const float* norm_w = (const float*)d_in[2];
  const float* norm_b = (const float*)d_in[3];
  const float* n1_w   = (const float*)d_in[4];
  const float* n1_b   = (const float*)d_in[5];
  const float* qkv_w  = (const float*)d_in[6];
  const float* qkv_b  = (const float*)d_in[7];
  const float* proj_w = (const float*)d_in[8];
  const float* proj_b = (const float*)d_in[9];
  const float* rpb    = (const float*)d_in[10];
  const float* n2_w   = (const float*)d_in[11];
  const float* n2_b   = (const float*)d_in[12];
  const float* fc1_w  = (const float*)d_in[13];
  const float* fc1_b  = (const float*)d_in[14];
  const float* fc2_w  = (const float*)d_in[15];
  const float* fc2_b  = (const float*)d_in[16];
  const float* conv_w = (const float*)d_in[17];
  const float* conv_b = (const float*)d_in[18];

  char* ws = (char*)d_ws;
  // workspace layout (all 256-aligned)
  const size_t OFF_Z     = 0;                               // f32 NTOKx384  (z -> z2 in place)
  const size_t OFF_WIN   = OFF_Z     + (size_t)NTOK * C2 * 4;    // bf16 NTOKx384 (zn_win -> attn_out)
  const size_t OFF_QKV   = OFF_WIN   + (size_t)NTOK * C2 * 2;    // bf16 NTOKx1152 (qkv -> reused as HIN)
  const size_t OFF_HMID  = OFF_QKV   + (size_t)NTOK * 1152 * 2;  // bf16 NTOKx1536
  const size_t OFF_Z3    = OFF_HMID  + (size_t)NTOK * MLPH * 2;  // bf16 NTOKx384
  const size_t OFF_QKVW  = OFF_Z3    + (size_t)NTOK * C2 * 2;
  const size_t OFF_PROJW = OFF_QKVW  + (size_t)1152 * 384 * 2;
  const size_t OFF_FC1W  = OFF_PROJW + (size_t)384 * 384 * 2;
  const size_t OFF_FC2W  = OFF_FC1W  + (size_t)1536 * 384 * 2;
  const size_t OFF_CONVW = OFF_FC2W  + (size_t)384 * 1536 * 2;   // end ~273.1 MB

  float*  Z     = (float*)(ws + OFF_Z);
  bf16_t* WIN   = (bf16_t*)(ws + OFF_WIN);
  bf16_t* QKV   = (bf16_t*)(ws + OFF_QKV);
  bf16_t* HIN   = (bf16_t*)(ws + OFF_QKV);   // reuse: qkv dead after attention
  bf16_t* HMID  = (bf16_t*)(ws + OFF_HMID);
  bf16_t* Z3    = (bf16_t*)(ws + OFF_Z3);
  bf16_t* QKVW  = (bf16_t*)(ws + OFF_QKVW);
  bf16_t* PROJW = (bf16_t*)(ws + OFF_PROJW);
  bf16_t* FC1W  = (bf16_t*)(ws + OFF_FC1W);
  bf16_t* FC2W  = (bf16_t*)(ws + OFF_FC2W);
  bf16_t* CONVW = (bf16_t*)(ws + OFF_CONVW);

  // weight converts (f32 -> bf16, (N,K) row-major kept as-is)
  cvt_f32_bf16<<<(1152 * 384 + 255) / 256, 256, 0, stream>>>(qkv_w, QKVW, 1152 * 384);
  cvt_f32_bf16<<<(384 * 384 + 255) / 256, 256, 0, stream>>>(proj_w, PROJW, 384 * 384);
  cvt_f32_bf16<<<(1536 * 384 + 255) / 256, 256, 0, stream>>>(fc1_w, FC1W, 1536 * 384);
  cvt_f32_bf16<<<(384 * 1536 + 255) / 256, 256, 0, stream>>>(fc2_w, FC2W, 384 * 1536);
  conv_repack<<<(192 * CONVK + 255) / 256, 256, 0, stream>>>(conv_w, CONVW);

  const int LNBLOCKS = NTOK / 8;  // 3969, exact
  concat_ln_kernel<<<LNBLOCKS, 256, 0, stream>>>(x, g, norm_w, norm_b, Z);
  ln_bf16_kernel<<<LNBLOCKS, 256, 0, stream>>>(Z, n1_w, n1_b, WIN, 1);

  const int MB = (NTOK + 127) / 128;  // 249
  // QKV: (31752 x 1152) = WIN(31752x384) x qkv_w^T
  gemm_bf16_kernel<<<dim3(MB, 1152 / 64), 256, 0, stream>>>(
      WIN, QKVW, qkv_b, QKV, nullptr, NTOK, 1152, 384, 0);

  attn_kernel<<<NWIN * HEADS, 128, 0, stream>>>(QKV, rpb, WIN);

  // proj + window-reverse + unshift + residual (in-place into Z -> z2)
  gemm_bf16_kernel<<<dim3(MB, 384 / 64), 256, 0, stream>>>(
      WIN, PROJW, proj_b, nullptr, Z, NTOK, 384, 384, 2);

  ln_bf16_kernel<<<LNBLOCKS, 256, 0, stream>>>(Z, n2_w, n2_b, HIN, 0);

  gemm_bf16_kernel<<<dim3(MB, 1536 / 64), 256, 0, stream>>>(
      HIN, FC1W, fc1_b, HMID, nullptr, NTOK, 1536, 384, 1);

  gemm_bf16_kernel<<<dim3(MB, 384 / 64), 256, 0, stream>>>(
      HMID, FC2W, fc2_b, Z3, Z, NTOK, 384, 1536, 3);

  conv_kernel<<<dim3((CONVM + 127) / 128, 192 / 64), 256, 0, stream>>>(
      Z3, CONVW, conv_b, x, (float*)d_out);
}
